// AttentionBlock_38594576121909
// MI455X (gfx1250) — matrix operand order repaired and checked by reading, not compiled, not run on hardware
//
#include <hip/hip_runtime.h>
#include <hip/hip_bf16.h>

// ---------------------------------------------------------------------------
// Problem constants (reference: b=16, C=128, 32x32 spatial, 4 heads, d_k=32)
// ---------------------------------------------------------------------------
#define BATCH 16
#define CCH   128
#define NTOK  1024
#define NH    4
#define DK    32
#define NPROJ 384                     // NH * DK * 3
#define SCALE 0.17677669529663687f    // 32^-0.5

typedef __attribute__((ext_vector_type(16))) __bf16 v16bf;
typedef __attribute__((ext_vector_type(8)))  __bf16 v8bf;
typedef __attribute__((ext_vector_type(8)))  float  v8f;
typedef __attribute__((ext_vector_type(4)))  float  v4f;

__device__ inline v8f zero8() {
    v8f z;
#pragma unroll
    for (int i = 0; i < 8; ++i) z[i] = 0.0f;
    return z;
}

__device__ inline v8f wmma_bf16(v16bf a, v16bf b, v8f c) {
    // D = A(16x32) * B(32x16) + C, f32 accumulate
    return __builtin_amdgcn_wmma_f32_16x16x32_bf16(
        false, a, false, b, (short)0, c, false, false);
}

// Build a 16-element bf16 fragment from two 16-byte chunks.
__device__ inline v16bf frag_from2(const __bf16* p0, const __bf16* p1) {
    v8bf lo = *(const v8bf*)p0;
    v8bf hi = *(const v8bf*)p1;
    v16bf r;
#pragma unroll
    for (int t = 0; t < 8; ++t) { r[t] = lo[t]; r[t + 8] = hi[t]; }
    return r;
}

// A-matrix fragment (16 rows x K=32), source row-major with leading dim ldr.
// ISA layout: lanes 0-15 row=lane, K 0..7 & 16..23 ; lanes 16-31 row=lane-16,
// K 8..15 & 24..31.
__device__ inline v16bf load_A_bf16(const __bf16* tile, int ldr, int lane) {
    int row = lane & 15;
    int ko  = (lane >> 4) << 3;            // 0 or 8
    const __bf16* p = tile + row * ldr + ko;
    return frag_from2(p, p + 16);
}

// B-matrix fragment (K=32 x 16 cols), source stored as N rows x K cols
// (i.e. the transpose of B) row-major with leading dim ldr.
// Assumed layout: lane holds column (lane&15); lanes 0-15 K 0..7 & 16..23,
// lanes 16-31 K 8..15 & 24..31, packed pairs per VGPR.
__device__ inline v16bf load_B_nk(const __bf16* nk, int ldr, int lane) {
    int col = lane & 15;
    int ko  = (lane >> 4) << 3;            // 0 or 8
    const __bf16* p = nk + col * ldr + ko; // K ko..ko+7 and ko+16..ko+23
    return frag_from2(p, p + 16);
}

// ---------------------------------------------------------------------------
// Kernel 0: one-shot weight conversion f32 -> bf16 (same row-major layout).
// Wp: 384x128, Wo: 128x128. Hot loops then load pure bf16 fragments.
// ---------------------------------------------------------------------------
__global__ __launch_bounds__(256) void k_cvt_w(const float* __restrict__ Wp,
                                               const float* __restrict__ Wo,
                                               __bf16* __restrict__ Wpb,
                                               __bf16* __restrict__ Wob) {
    int idx = blockIdx.x * 256 + threadIdx.x;       // 65536 total
    const int np = NPROJ * CCH;                     // 49152
    if (idx < np) Wpb[idx] = (__bf16)Wp[idx];
    else          Wob[idx - np] = (__bf16)Wo[idx - np];
}

// ---------------------------------------------------------------------------
// Kernel 1: x (b,c,n) f32  ->  xt (b,n,c) bf16   (LDS tile transpose)
// ---------------------------------------------------------------------------
__global__ __launch_bounds__(256) void k_transpose(const float* __restrict__ x,
                                                   __bf16* __restrict__ xt) {
    __shared__ float t[32][33];
    int i0 = blockIdx.x * 32;
    int c0 = blockIdx.y * 32;
    int b  = blockIdx.z;
#pragma unroll
    for (int r = 0; r < 4; ++r) {
        int c = c0 + threadIdx.y + r * 8;
        t[threadIdx.y + r * 8][threadIdx.x] =
            x[((size_t)b * CCH + c) * NTOK + i0 + threadIdx.x];
    }
    __syncthreads();
#pragma unroll
    for (int r = 0; r < 4; ++r) {
        int i = i0 + threadIdx.y + r * 8;
        xt[((size_t)b * NTOK + i) * CCH + c0 + threadIdx.x] =
            (__bf16)t[threadIdx.x][threadIdx.y + r * 8];
    }
}

// ---------------------------------------------------------------------------
// Kernel 2: qkv = xt @ Wp^T + bp  -> Q,K row-major [b,h,n,32] bf16,
//                                    V transposed  [b,h,32,n] bf16
// One wave per (16 i-rows x 16 o-cols) tile; K-loop over c in steps of 32.
// o-tile mapping: per head 96 outputs = 6 tiles: [q0 q1 k0 k1 v0 v1]
// ---------------------------------------------------------------------------
__global__ __launch_bounds__(128) void k_qkv(const __bf16* __restrict__ xt,
                                             const __bf16* __restrict__ Wpb,
                                             const float* __restrict__ bp,
                                             __bf16* __restrict__ Qb,
                                             __bf16* __restrict__ Kb,
                                             __bf16* __restrict__ Vt) {
    int lane = threadIdx.x & 31;
    int wid  = threadIdx.x >> 5;
    int i0   = (blockIdx.x * 4 + wid) * 16;
    int ot   = blockIdx.y;             // 0..23
    int b    = blockIdx.z;
    int o0   = ot * 16;

    v8f acc = zero8();
#pragma unroll
    for (int c0 = 0; c0 < CCH; c0 += 32) {
        v16bf a = load_A_bf16(xt + ((size_t)b * NTOK + i0) * CCH + c0, CCH, lane);
        v16bf w = load_B_nk(Wpb + (size_t)o0 * CCH + c0, CCH, lane);
        acc = wmma_bf16(a, w, acc);
    }
    float bias = bp[o0 + (lane & 15)];
#pragma unroll
    for (int r = 0; r < 8; ++r) acc[r] += bias;

    int h     = ot / 6;
    int sub   = ot % 6;
    int dblk  = sub & 1;               // 0: d 0..15, 1: d 16..31
    int which = sub >> 1;              // 0=q 1=k 2=v
    int iBase = i0 + ((lane >> 4) << 3);

    if (which < 2) {
        __bf16* dst = (which == 0 ? Qb : Kb) +
                      (((size_t)(b * NH + h) * NTOK) + iBase) * DK +
                      dblk * 16 + (lane & 15);
#pragma unroll
        for (int r = 0; r < 8; ++r) dst[(size_t)r * DK] = (__bf16)acc[r];
    } else {
        int d = dblk * 16 + (lane & 15);
        __bf16* dst = Vt + ((size_t)(b * NH + h) * DK + d) * NTOK + iBase;
        v8bf pk;
#pragma unroll
        for (int r = 0; r < 8; ++r) pk[r] = (__bf16)acc[r];
        *(v8bf*)dst = pk;              // 8 consecutive i -> one b128 store
    }
}

// ---------------------------------------------------------------------------
// Kernel 3: fused attention (softmax over HEADS axis, per reference bug)
//           + output projection + bias + residual, f32 output in (b,c,n).
// One wave owns a 16-row i-tile for all 4 heads.
// ---------------------------------------------------------------------------
__global__ __launch_bounds__(128) void k_attn(const __bf16* __restrict__ Qb,
                                              const __bf16* __restrict__ Kb,
                                              const __bf16* __restrict__ Vt,
                                              const __bf16* __restrict__ Wob,
                                              const float* __restrict__ bo,
                                              const float* __restrict__ x,
                                              float* __restrict__ out) {
    __shared__ __align__(16) __bf16 ldsP[4][NH][16 * 32];   // per-wave P staging
    __shared__ __align__(16) __bf16 ldsO[4][16 * 128];      // per-wave O staging

    int lane = threadIdx.x & 31;
    int wid  = threadIdx.x >> 5;
    int i0   = (blockIdx.x * 4 + wid) * 16;
    int b    = blockIdx.y;

    int laneHi = lane >> 4;
    int laneN  = lane & 15;

    // Persistent Q fragments (A layout) for all heads
    v16bf qf[NH];
#pragma unroll
    for (int h = 0; h < NH; ++h)
        qf[h] = load_A_bf16(Qb + ((size_t)(b * NH + h) * NTOK + i0) * DK, DK, lane);

    v8f of[NH][2];
#pragma unroll
    for (int h = 0; h < NH; ++h) { of[h][0] = zero8(); of[h][1] = zero8(); }

    for (int j0 = 0; j0 < NTOK; j0 += 32) {
        // ---- scores: S_h = Q_h K_h^T for two 16-wide j sub-tiles ----
        v8f s[NH][2];
#pragma unroll
        for (int h = 0; h < NH; ++h) {
            const __bf16* kb = Kb + ((size_t)(b * NH + h) * NTOK + j0) * DK;
            v16bf k0 = load_B_nk(kb, DK, lane);
            v16bf k1 = load_B_nk(kb + 16 * DK, DK, lane);
            s[h][0] = wmma_bf16(qf[h], k0, zero8());
            s[h][1] = wmma_bf16(qf[h], k1, zero8());
        }
        // ---- softmax across the 4 heads (elementwise in i,j) ----
#pragma unroll
        for (int sub = 0; sub < 2; ++sub) {
#pragma unroll
            for (int r = 0; r < 8; ++r) {
                float a0 = s[0][sub][r] * SCALE;
                float a1 = s[1][sub][r] * SCALE;
                float a2 = s[2][sub][r] * SCALE;
                float a3 = s[3][sub][r] * SCALE;
                float m  = fmaxf(fmaxf(a0, a1), fmaxf(a2, a3));
                float e0 = __expf(a0 - m), e1 = __expf(a1 - m);
                float e2 = __expf(a2 - m), e3 = __expf(a3 - m);
                float inv = 1.0f / (e0 + e1 + e2 + e3);
                int mrow = r + (laneHi << 3);
                int jcol = sub * 16 + laneN;
                ldsP[wid][0][mrow * 32 + jcol] = (__bf16)(e0 * inv);
                ldsP[wid][1][mrow * 32 + jcol] = (__bf16)(e1 * inv);
                ldsP[wid][2][mrow * 32 + jcol] = (__bf16)(e2 * inv);
                ldsP[wid][3][mrow * 32 + jcol] = (__bf16)(e3 * inv);
            }
        }
        // ---- O_h += P_h @ V_h  (K = 32 j's, two 16-wide d halves) ----
#pragma unroll
        for (int h = 0; h < NH; ++h) {
            v16bf pf = load_A_bf16(ldsP[wid][h], 32, lane);
            const __bf16* vb = Vt + ((size_t)(b * NH + h) * DK) * NTOK + j0;
            v16bf v0 = load_B_nk(vb, NTOK, lane);                     // d 0..15
            v16bf v1 = load_B_nk(vb + (size_t)16 * NTOK, NTOK, lane); // d 16..31
            of[h][0] = wmma_bf16(pf, v0, of[h][0]);
            of[h][1] = wmma_bf16(pf, v1, of[h][1]);
        }
    }

    // ---- stage O (16 x 128, [h*32+d] concat) into LDS for reuse as A ----
#pragma unroll
    for (int h = 0; h < NH; ++h)
#pragma unroll
        for (int dblk = 0; dblk < 2; ++dblk)
#pragma unroll
            for (int r = 0; r < 8; ++r) {
                int mrow = r + (laneHi << 3);
                int col  = h * 32 + dblk * 16 + laneN;
                ldsO[wid][mrow * 128 + col] = (__bf16)of[h][dblk][r];
            }

    // Hoisted A fragments over the 4 K-chunks of the output projection
    v16bf oa[4];
#pragma unroll
    for (int kc = 0; kc < 4; ++kc)
        oa[kc] = load_A_bf16(ldsO[wid] + kc * 32, 128, lane);

    // ---- res = O @ Wo^T + bo + xf ; store as out[b][c][i] ----
    int iBase = i0 + (laneHi << 3);
#pragma unroll
    for (int ct = 0; ct < 8; ++ct) {
        int c0 = ct * 16;
        v8f acc = zero8();
#pragma unroll
        for (int kc = 0; kc < 4; ++kc) {
            v16bf wb = load_B_nk(Wob + (size_t)c0 * 128 + kc * 32, 128, lane);
            acc = wmma_bf16(oa[kc], wb, acc);
        }
        int c = c0 + laneN;
        float bias = bo[c];
        const float* xrow = x + ((size_t)b * CCH + c) * NTOK + iBase;
        v4f x0 = *(const v4f*)xrow;
        v4f x1 = *(const v4f*)(xrow + 4);
        v4f o0v, o1v;
#pragma unroll
        for (int t = 0; t < 4; ++t) {
            o0v[t] = acc[t]     + bias + x0[t];
            o1v[t] = acc[t + 4] + bias + x1[t];
        }
        float* orow = out + ((size_t)b * CCH + c) * NTOK + iBase;
        *(v4f*)orow       = o0v;
        *(v4f*)(orow + 4) = o1v;
    }
}

// ---------------------------------------------------------------------------
// Launch
// ---------------------------------------------------------------------------
extern "C" void kernel_launch(void* const* d_in, const int* in_sizes, int n_in,
                              void* d_out, int out_size, void* d_ws, size_t ws_size,
                              hipStream_t stream) {
    const float* x  = (const float*)d_in[0];
    const float* Wp = (const float*)d_in[1];
    const float* bp = (const float*)d_in[2];
    const float* Wo = (const float*)d_in[3];
    const float* bo = (const float*)d_in[4];
    float* out = (float*)d_out;

    char* ws = (char*)d_ws;
    __bf16* xt  = (__bf16*)(ws);                                // 4 MiB
    __bf16* Qb  = (__bf16*)(ws + (size_t)4  * 1024 * 1024);     // 4 MiB
    __bf16* Kb  = (__bf16*)(ws + (size_t)8  * 1024 * 1024);     // 4 MiB
    __bf16* Vt  = (__bf16*)(ws + (size_t)12 * 1024 * 1024);     // 4 MiB
    __bf16* Wpb = (__bf16*)(ws + (size_t)16 * 1024 * 1024);     // 96 KiB
    __bf16* Wob = (__bf16*)(ws + (size_t)16 * 1024 * 1024 + 96 * 1024); // 32 KiB

    k_cvt_w<<<dim3((NPROJ * CCH + CCH * CCH) / 256), 256, 0, stream>>>(Wp, Wo, Wpb, Wob);
    k_transpose<<<dim3(NTOK / 32, CCH / 32, BATCH), dim3(32, 8), 0, stream>>>(x, xt);
    k_qkv<<<dim3(NTOK / 64, 24, BATCH), 128, 0, stream>>>(xt, Wpb, bp, Qb, Kb, Vt);
    k_attn<<<dim3(NTOK / 64, BATCH), 128, 0, stream>>>(Qb, Kb, Vt, Wob, bo, x, out);
}
